// ActorNetwork_35081292874064
// MI455X (gfx1250) — matrix operand order repaired and checked by reading, not compiled, not run on hardware
//
#include <hip/hip_runtime.h>
#include <hip/hip_bf16.h>
#include <math.h>

typedef __attribute__((ext_vector_type(16))) _Float16 v16h;
typedef __attribute__((ext_vector_type(8)))  _Float16 v8h;
typedef __attribute__((ext_vector_type(4)))  _Float16 v4h;
typedef __attribute__((ext_vector_type(8)))  float    v8f;
typedef __attribute__((ext_vector_type(4)))  int      v4i;

#define N_NODES 50000
#define N_EDGES 800000
#define N_AGENTS 8192
#define HID 256

// ---------------------------------------------------------------------------
// CDNA5 async global->LDS copy (ASYNCcnt path), guarded for toolchain support
// ---------------------------------------------------------------------------
__device__ __forceinline__ void async_copy_b128(void* lds, const void* gptr) {
#if __has_builtin(__builtin_amdgcn_global_load_async_to_lds_b128)
    __builtin_amdgcn_global_load_async_to_lds_b128(
        (__attribute__((address_space(1))) v4i*)(void*)gptr,
        (__attribute__((address_space(3))) v4i*)(void*)lds, 0, 0);
#else
    *(v8h*)lds = *(const v8h*)gptr;      // fallback: b128 load + b128 ds store
#endif
}

__device__ __forceinline__ void wait_async() {
#if __has_builtin(__builtin_amdgcn_s_wait_asynccnt)
    __builtin_amdgcn_s_wait_asynccnt(0);
#elif __has_builtin(__builtin_amdgcn_global_load_async_to_lds_b128)
    asm volatile("s_wait_asynccnt 0x0" ::: "memory");
#endif
}

// ---------------------------------------------------------------------------
// Small elementwise / graph kernels
// ---------------------------------------------------------------------------
__global__ __launch_bounds__(256) void init_deg(float* deg, int n) {
    int i = blockIdx.x * 256 + threadIdx.x;
    if (i < n) deg[i] = 1.0f;               // self-loop contributes 1
}

__global__ __launch_bounds__(256) void deg_count(const int* __restrict__ dst,
                                                 float* deg, int E) {
    int i = blockIdx.x * 256 + threadIdx.x;
    if (i < E) atomicAdd(&deg[dst[i]], 1.0f);
}

__global__ __launch_bounds__(256) void deg_to_dis(float* deg, int n) {
    int i = blockIdx.x * 256 + threadIdx.x;
    if (i < n) deg[i] = rsqrtf(deg[i]);     // deg >= 1 always (self-loops)
}

// Bt[n][k] = (f16) B[k][n]  -- pre-transpose weights once (tiny matrices)
__global__ __launch_bounds__(256) void conv_tr_f16(const float* __restrict__ B,
                                                   _Float16* __restrict__ Bt,
                                                   int K, int N) {
    int i = blockIdx.x * 256 + threadIdx.x;
    if (i < K * N) {
        int k = i / N, n = i % N;
        Bt[(size_t)n * K + k] = (_Float16)B[i];
    }
}

// h[n][c] = b_gcn[c] + xw[n][c] * dis[n]^2   (bias + self-loop term)
__global__ __launch_bounds__(256) void h_init(const float* __restrict__ xw,
                                              const float* __restrict__ dis,
                                              const float* __restrict__ bgcn,
                                              float* __restrict__ h, int n) {
    int i = blockIdx.x * 256 + threadIdx.x;
    int node = i >> 8, c = i & 255;
    if (node < n) {
        float d = dis[node];
        h[i] = bgcn[c] + xw[i] * d * d;
    }
}

// one wave per edge; lane-strided channels -> coalesced reads, L2 atomics
__global__ __launch_bounds__(256) void scatter_edges(const int* __restrict__ src,
                                                     const int* __restrict__ dst,
                                                     const float* __restrict__ xw,
                                                     const float* __restrict__ dis,
                                                     float* __restrict__ h, int E) {
    int lane = threadIdx.x & 31;
    int e = (blockIdx.x * blockDim.x + threadIdx.x) >> 5;   // wave-uniform
    if (e >= E) return;
    int s = src[e], d = dst[e];
    float norm = dis[s] * dis[d];
    const float* xs = xw + (size_t)s * HID;
    float* hd = h + (size_t)d * HID;
    for (int c = lane; c < HID; c += 32)
        atomicAdd(&hd[c], xs[c] * norm);
}

__global__ __launch_bounds__(256) void gather_relu(const float* __restrict__ h,
                                                   const int* __restrict__ aidx,
                                                   float* __restrict__ hA) {
    int i = blockIdx.x * 256 + threadIdx.x;
    int a = i >> 8, c = i & 255;
    int node = aidx[a];
    hA[i] = fmaxf(h[(size_t)node * HID + c], 0.0f);
}

// ---------------------------------------------------------------------------
// LayerNorm + ReLU, one block (256 threads) per row
// ---------------------------------------------------------------------------
__global__ __launch_bounds__(256) void ln_relu(float* __restrict__ h,
                                               const float* __restrict__ g,
                                               const float* __restrict__ be,
                                               int D) {
    __shared__ float ssum[256];
    __shared__ float ssq[256];
    float* p = h + (size_t)blockIdx.x * D;
    float s = 0.f, q = 0.f;
    for (int c = threadIdx.x; c < D; c += 256) {
        float v = p[c];
        s += v; q += v * v;
    }
    ssum[threadIdx.x] = s; ssq[threadIdx.x] = q;
    __syncthreads();
    for (int off = 128; off > 0; off >>= 1) {
        if (threadIdx.x < off) {
            ssum[threadIdx.x] += ssum[threadIdx.x + off];
            ssq[threadIdx.x]  += ssq[threadIdx.x + off];
        }
        __syncthreads();
    }
    float mu  = ssum[0] / (float)D;
    float var = ssq[0] / (float)D - mu * mu;
    float rs  = rsqrtf(var + 1e-5f);
    for (int c = threadIdx.x; c < D; c += 256) {
        float v = (p[c] - mu) * rs * g[c] + be[c];
        p[c] = fmaxf(v, 0.0f);
    }
}

// ---------------------------------------------------------------------------
// WMMA f16 GEMM: C[M,N] = A[M,K] * Bt^T (+bias), Bt = f16 [N][K] pre-transposed.
// Block = 128 threads = 4 waves; block tile 64x64; K step 64 (8 WMMA/wave/step).
// A: f32->f16 packed b64 stores, row-clamped loads (no exec divergence).
// B: contiguous 16B async copies global->LDS (ASYNCcnt), no VGPR round-trip.
// ---------------------------------------------------------------------------
__global__ __launch_bounds__(128)
void gemm_f16wmma(const float* __restrict__ A, const _Float16* __restrict__ Bt,
                  float* __restrict__ C, const float* __restrict__ bias,
                  int M, int N, int K, int relu) {
    __shared__ _Float16 sA[64][72];   // 144B row stride (16B-aligned chunks)
    __shared__ _Float16 sB[64][72];   // [n][k]
    int t = threadIdx.x;
    int lane = t & 31;
    int wave = t >> 5;
    int mBlk = blockIdx.y * 64;
    int nBlk = blockIdx.x * 64;

    v8f acc[4] = {};

    for (int k0 = 0; k0 < K; k0 += 64) {
        // ---- stage A tile 64x64 (f32 -> f16, packed b64 stores) ----
        for (int i = 0; i < 8; ++i) {
            int idx = t + i * 128;            // 0..1023
            int r = idx >> 4;                 // 0..63
            int c4 = (idx & 15) << 2;         // 0..60
            int gr = mBlk + r; gr = gr < M ? gr : M - 1;   // clamp, no branch
            float4 v = *(const float4*)(A + (size_t)gr * K + k0 + c4);
            v4h hv;
            hv[0] = (_Float16)v.x; hv[1] = (_Float16)v.y;
            hv[2] = (_Float16)v.z; hv[3] = (_Float16)v.w;
            *(v4h*)&sA[r][c4] = hv;
        }
        // ---- stage B tile: 64 rows x 64 halfs, async 16B copies ----
        for (int i = 0; i < 4; ++i) {
            int idx = t + i * 128;            // 0..511
            int n = idx >> 3;                 // 0..63
            int ch = (idx & 7) << 3;          // 0..56 (halfs)
            async_copy_b128(&sB[n][ch],
                            Bt + (size_t)(nBlk + n) * K + k0 + ch);
        }
        wait_async();
        __syncthreads();

        // ---- fragments + 8 WMMAs per wave ----
        int m = wave * 16 + (lane & 15);
        for (int s = 0; s < 2; ++s) {
            int kA = s * 32 + (lane >> 4) * 8;
            union { v16h v; v8h h[2]; } a;
            a.h[0] = *(const v8h*)&sA[m][kA];
            a.h[1] = *(const v8h*)&sA[m][kA + 16];
            int kB = s * 32 + (lane >> 4) * 8;
            for (int nt = 0; nt < 4; ++nt) {
                int n = nt * 16 + (lane & 15);
                union { v16h v; v8h h[2]; } b;
                b.h[0] = *(const v8h*)&sB[n][kB];
                b.h[1] = *(const v8h*)&sB[n][kB + 16];
                acc[nt] = __builtin_amdgcn_wmma_f32_16x16x32_f16(
                    false, a.v, false, b.v, (short)0, acc[nt], false, false);
            }
        }
        __syncthreads();
    }

    // ---- epilogue: C/D layout: VGPR r -> row r + 8*(lane/16), col = lane%16
    int mRow = mBlk + wave * 16 + (lane >> 4) * 8;
    int col0 = nBlk + (lane & 15);
    for (int nt = 0; nt < 4; ++nt) {
        int col = col0 + nt * 16;
        float bv = bias ? bias[col] : 0.0f;
        for (int r = 0; r < 8; ++r) {
            int row = mRow + r;
            if (row < M) {
                float v = acc[nt][r] + bv;
                if (relu) v = fmaxf(v, 0.0f);
                C[(size_t)row * N + col] = v;
            }
        }
    }
}

// ---------------------------------------------------------------------------
// Head: out[8192,16] = sigmoid(t2[8192,512] @ Wmu + bmu), Wmu_t = f16 [16][512]
// One wave per 16x16 output tile, 16 chained WMMAs over K=512.
// ---------------------------------------------------------------------------
__global__ __launch_bounds__(128)
void head_sigmoid(const float* __restrict__ A, const _Float16* __restrict__ Wt,
                  const float* __restrict__ bmu, float* __restrict__ out) {
    int lane = threadIdx.x & 31;
    int wave = threadIdx.x >> 5;
    int mBase = blockIdx.x * 64 + wave * 16;
    int m = mBase + (lane & 15);
    const float* arow = A + (size_t)m * 512;
    int n = lane & 15;
    const _Float16* wrow = Wt + (size_t)n * 512;

    v8f acc = {};
    for (int k0 = 0; k0 < 512; k0 += 32) {
        int kA = k0 + (lane >> 4) * 8;
        union { v16h v; v8h h[2]; } a;
        for (int j = 0; j < 8; ++j) a.h[0][j] = (_Float16)arow[kA + j];
        for (int j = 0; j < 8; ++j) a.h[1][j] = (_Float16)arow[kA + 16 + j];
        int kB = k0 + (lane >> 4) * 8;
        union { v16h v; v8h h[2]; } b;
        b.h[0] = *(const v8h*)(wrow + kB);
        b.h[1] = *(const v8h*)(wrow + kB + 16);
        acc = __builtin_amdgcn_wmma_f32_16x16x32_f16(
            false, a.v, false, b.v, (short)0, acc, false, false);
    }
    int row = mBase + 8 * (lane >> 4);
    float bv = bmu[n];
    for (int r = 0; r < 8; ++r) {
        float v = acc[r] + bv;
        out[(size_t)(row + r) * 16 + n] = 1.0f / (1.0f + __expf(-v));
    }
}

// ---------------------------------------------------------------------------
extern "C" void kernel_launch(void* const* d_in, const int* in_sizes, int n_in,
                              void* d_out, int out_size, void* d_ws, size_t ws_size,
                              hipStream_t stream) {
    (void)in_sizes; (void)n_in; (void)out_size; (void)ws_size;
    const float* x    = (const float*)d_in[0];
    const int*   edge = (const int*)d_in[1];
    const int*   aidx = (const int*)d_in[2];
    const float* Wg   = (const float*)d_in[3];
    const float* bg   = (const float*)d_in[4];
    const float* W1   = (const float*)d_in[5];
    const float* b1   = (const float*)d_in[6];
    const float* g1   = (const float*)d_in[7];
    const float* be1  = (const float*)d_in[8];
    const float* W2   = (const float*)d_in[9];
    const float* b2   = (const float*)d_in[10];
    const float* g2   = (const float*)d_in[11];
    const float* be2  = (const float*)d_in[12];
    const float* Wmu  = (const float*)d_in[13];
    const float* bmu  = (const float*)d_in[14];
    float* out = (float*)d_out;

    // workspace layout (aliased; ~104.3 MB total)
    char* ws = (char*)d_ws;
    float* dis  = (float*)ws;                                  // 200,192 B
    float* bufA = (float*)(ws + 200192);                       // 51,200,000 B
    float* bufB = (float*)(ws + 200192 + 51200000);            // 51,200,000 B
    char*  wsc  = ws + 200192 + 2 * 51200000;                  // f16 weights
    _Float16* Wg_t  = (_Float16*)wsc;                          // [256][128]   65,536 B
    _Float16* W1_t  = (_Float16*)(wsc + 65536);                // [1024][256] 524,288 B
    _Float16* W2_t  = (_Float16*)(wsc + 65536 + 524288);       // [512][1024] 1,048,576 B
    _Float16* Wmu_t = (_Float16*)(wsc + 65536 + 524288 + 1048576); // [16][512] 16,384 B

    float* xw = bufA;                                          // [50000,256]
    float* h  = bufB;                                          // [50000,256]
    float* t1 = bufA;                                          // [8192,1024] (xw dead)
    float* hA = (float*)((char*)bufA + 33554432);              // [8192,256] after t1
    float* t2 = bufB;                                          // [8192,512]  (h dead)

    const int N = N_NODES, E = N_EDGES, A = N_AGENTS;
    const int* esrc = edge;
    const int* edst = edge + E;

    // pre-transpose + f16-convert all weights (tiny, one-time per launch)
    conv_tr_f16<<<(128 * 256 + 255) / 256, 256, 0, stream>>>(Wg, Wg_t, 128, 256);
    conv_tr_f16<<<(256 * 1024 + 255) / 256, 256, 0, stream>>>(W1, W1_t, 256, 1024);
    conv_tr_f16<<<(1024 * 512 + 255) / 256, 256, 0, stream>>>(W2, W2_t, 1024, 512);
    conv_tr_f16<<<(512 * 16 + 255) / 256, 256, 0, stream>>>(Wmu, Wmu_t, 512, 16);

    // degree -> dis = rsqrt(deg)
    init_deg  <<<(N + 255) / 256, 256, 0, stream>>>(dis, N);
    deg_count <<<(E + 255) / 256, 256, 0, stream>>>(edst, dis, E);
    deg_to_dis<<<(N + 255) / 256, 256, 0, stream>>>(dis, N);

    // xw = x @ W_gcn
    dim3 gx(HID / 64, (N + 63) / 64);
    gemm_f16wmma<<<gx, 128, 0, stream>>>(x, Wg_t, xw, nullptr, N, HID, 128, 0);

    // h = bias + self-loop + edge scatter
    h_init<<<N, 256, 0, stream>>>(xw, dis, bg, h, N);
    scatter_edges<<<(E + 7) / 8, 256, 0, stream>>>(esrc, edst, xw, dis, h, E);

    // agent gather + relu
    gather_relu<<<A, 256, 0, stream>>>(h, aidx, hA);

    // FC1 + LN + ReLU
    dim3 gf1(1024 / 64, A / 64);
    gemm_f16wmma<<<gf1, 128, 0, stream>>>(hA, W1_t, t1, b1, A, 1024, 256, 0);
    ln_relu<<<A, 256, 0, stream>>>(t1, g1, be1, 1024);

    // FC2 + LN + ReLU
    dim3 gf2(512 / 64, A / 64);
    gemm_f16wmma<<<gf2, 128, 0, stream>>>(t1, W2_t, t2, b2, A, 512, 1024, 0);
    ln_relu<<<A, 256, 0, stream>>>(t2, g2, be2, 512);

    // head + sigmoid
    head_sigmoid<<<A / 64, 128, 0, stream>>>(t2, Wmu_t, bmu, out);
}
